// MoELoraLinear_14070312862078
// MI455X (gfx1250) — matrix operand order repaired and checked by reading, not compiled, not run on hardware
//
#include <hip/hip_runtime.h>

// MoE-LoRA linear, collapsed to W_eff = W + sum_e gate_e * B_e @ A_e, then
// one big GEMM  out = x @ W_eff^T + bias  done with split-bf16 WMMA
// (hi*hi + hi*lo + lo*hi) for ~fp32 accuracy on CDNA5 tensor pipes.
// Operand tiles are staged into LDS by the Tensor Data Mover (TDM) with
// LDS row padding done by the DMA engine itself (D#.pad_*); wave 0 of each
// workgroup drives all four tile streams, double-buffered on TENSORcnt.
//
// Workspace layout (assumes ws_size >= 192MB + 256B):
//   [0, 32)       float gate[8]
//   [256 ...)     ushort Whi[4096*4096]  (bf16 bits, hi plane of W_eff)
//   +32MB         ushort Wlo[4096*4096]  (bf16 residual plane)
//   +64MB         ushort Xhi[8192*4096]
//   +128MB        ushort Xlo[8192*4096]

#define DDIM 4096
#define MDIM 8192
#define NUM_E 8
#define RE 8
#define ER 64              // NUM_E * RE
#define SCALING 0.25f      // 16/64

typedef unsigned short ushort_t;
typedef unsigned int   uint_t;
typedef unsigned long long ull_t;

typedef __attribute__((ext_vector_type(16))) __bf16 v16bf;
typedef __attribute__((ext_vector_type(8)))  float  v8f;
typedef __attribute__((ext_vector_type(4)))  unsigned int tdm_v4u;
typedef __attribute__((ext_vector_type(8)))  int tdm_v8i;
typedef __attribute__((ext_vector_type(4)))  int tdm_v4i;

#if defined(__has_builtin)
#  if __has_builtin(__builtin_amdgcn_tensor_load_to_lds)
#    define HAVE_TDM 1
#  endif
#endif
#ifndef HAVE_TDM
#  define HAVE_TDM 0
#endif
#if __has_include(<hip/amd_detail/amd_gfx1250_TDM.h>)
#  define TDM_6ARG 1
#else
#  define TDM_6ARG 0
#endif

__device__ __forceinline__ ushort_t f2bf(float f) {
    uint_t u = __builtin_bit_cast(uint_t, f);
    u += 0x7FFFu + ((u >> 16) & 1u);       // round-to-nearest-even
    return (ushort_t)(u >> 16);
}
__device__ __forceinline__ float bf2f(ushort_t h) {
    uint_t u = ((uint_t)h) << 16;
    return __builtin_bit_cast(float, u);
}

// ---------------------------------------------------------------- kernel 1
// Per-task router: logits = embed @ router_w^T ; softmax ; top-2 ; gate.
__global__ void k_router(const float* __restrict__ embed,
                         const float* __restrict__ rw,
                         float* __restrict__ gate) {
    __shared__ float logits[NUM_E];
    int e = threadIdx.x >> 5;
    int lane = threadIdx.x & 31;
    float p = 0.f;
    for (int d = lane; d < DDIM; d += 32)
        p += embed[d] * rw[e * DDIM + d];
    #pragma unroll
    for (int off = 16; off > 0; off >>= 1)
        p += __shfl_down(p, off, 32);
    if (lane == 0) logits[e] = p;
    __syncthreads();
    if (threadIdx.x == 0) {
        float mx = logits[0];
        for (int i = 1; i < NUM_E; i++) mx = fmaxf(mx, logits[i]);
        float pr[NUM_E], s = 0.f;
        for (int i = 0; i < NUM_E; i++) { pr[i] = __expf(logits[i] - mx); s += pr[i]; }
        float inv = 1.f / s;
        for (int i = 0; i < NUM_E; i++) pr[i] *= inv;
        int i1 = 0;
        for (int i = 1; i < NUM_E; i++) if (pr[i] > pr[i1]) i1 = i;
        int i2 = (i1 == 0) ? 1 : 0;
        for (int i = 0; i < NUM_E; i++) if (i != i1 && pr[i] > pr[i2]) i2 = i;
        for (int i = 0; i < NUM_E; i++)
            gate[i] = (i == i1 || i == i2) ? pr[i] * SCALING : 0.f;
    }
}

// ---------------------------------------------------------------- kernel 2
// W_eff[o,d] = W[o,d] + sum_{er} (gate[e]*B[e,o,r]) * A[er,d]
// Output directly as bf16 hi/lo planes. Tile: 64(o) x 64(d), 256 threads.
__global__ __launch_bounds__(256)
void k_weff(const float* __restrict__ W,
            const float* __restrict__ Bm,     // [E, D, RE]
            const float* __restrict__ Am,     // [E*RE, D] (A flattened)
            const float* __restrict__ gate,
            ushort_t* __restrict__ whi,
            ushort_t* __restrict__ wlo) {
    __shared__ float sA[ER][64 + 1];   // [er][d_local]
    __shared__ float sB[64][ER + 1];   // [o_local][er], gate pre-applied
    const int d0 = blockIdx.x * 64;
    const int o0 = blockIdx.y * 64;
    const int t  = threadIdx.x;

    for (int i = t; i < ER * 64; i += 256) {
        int er = i >> 6, dl = i & 63;
        sA[er][dl] = Am[(size_t)er * DDIM + d0 + dl];
    }
    for (int i = t; i < 64 * ER; i += 256) {
        int ol = i >> 6, er = i & 63;
        int e = er >> 3, r = er & 7;
        sB[ol][er] = gate[e] * Bm[((size_t)e * DDIM + (o0 + ol)) * RE + r];
    }
    __syncthreads();

    const int dl = t & 63;
    const int og = t >> 6;          // 0..3, each handles 16 o-rows
    float areg[ER];
    #pragma unroll
    for (int k = 0; k < ER; k++) areg[k] = sA[k][dl];

    #pragma unroll
    for (int oi = 0; oi < 16; oi++) {
        const int ol = og * 16 + oi;
        float s = 0.f;
        #pragma unroll
        for (int k = 0; k < ER; k++) s = fmaf(sB[ol][k], areg[k], s);
        const size_t idx = (size_t)(o0 + ol) * DDIM + (d0 + dl);
        const float w = W[idx] + s;
        const ushort_t hb = f2bf(w);
        const ushort_t lb = f2bf(w - bf2f(hb));
        whi[idx] = hb;
        wlo[idx] = lb;
    }
}

// ---------------------------------------------------------------- kernel 2b
// Split x into bf16 hi/lo planes (one-time, memory-bound pre-pass).
__global__ __launch_bounds__(256)
void k_xsplit(const float* __restrict__ X,
              ushort_t* __restrict__ xhi, ushort_t* __restrict__ xlo) {
    const size_t base = ((size_t)blockIdx.x * 256 + threadIdx.x) * 8;
    const float4 a = *(const float4*)(X + base);
    const float4 b = *(const float4*)(X + base + 4);
    const float v[8] = {a.x, a.y, a.z, a.w, b.x, b.y, b.z, b.w};
    uint4 hp, lp;
    uint_t* hq = (uint_t*)&hp;
    uint_t* lq = (uint_t*)&lp;
    #pragma unroll
    for (int i = 0; i < 4; i++) {
        ushort_t h0 = f2bf(v[2 * i]),     h1 = f2bf(v[2 * i + 1]);
        ushort_t l0 = f2bf(v[2 * i] - bf2f(h0));
        ushort_t l1 = f2bf(v[2 * i + 1] - bf2f(h1));
        hq[i] = (uint_t)h0 | ((uint_t)h1 << 16);
        lq[i] = (uint_t)l0 | ((uint_t)l1 << 16);
    }
    *(uint4*)(xhi + base) = hp;
    *(uint4*)(xlo + base) = lp;
}

// ---------------------------------------------------------------- kernel 3
// out[m,n] = sum_k x[m,k]*W_eff[n,k] + bias[n], split-bf16 WMMA,
// operand tiles staged by the Tensor Data Mover (wave 0 drives all 4 streams).
#define MT 128
#define NT 128
#define KT 32
#define NK (DDIM / KT)
#define LROW 40    // LDS row stride in bf16 elements (80B, 16B aligned)

__device__ __forceinline__ v16bf ldfrag(const ushort_t* __restrict__ s,
                                        int row, int c0, int c1) {
    union { uint4 q[2]; v16bf v; } u;
    u.q[0] = *(const uint4*)(s + row * LROW + c0);
    u.q[1] = *(const uint4*)(s + row * LROW + c1);
    return u.v;
}

#if HAVE_TDM
// D# group1 for a 32(K) x 128(rows) bf16 tile out of a [rows, 4096] bf16
// tensor, with DMA-side LDS padding: 16B pad after every 64B row
// -> 80B LDS row stride (matches LROW).
__device__ __forceinline__ tdm_v8i tdm_make_g1() {
    tdm_v8i g;
    g[0] = (1 << 16)        // data_size = 2 bytes
         | (1 << 20)        // pad_enable
         | (3 << 22)        // pad_interval: 16 DWORDs = 64B
         | (3 << 25);       // pad_amount:   4 DWORDs = 16B
    g[1] = (int)(4096u << 16);   // tensor_dim0 = 4096 (bits 79:48)
    g[2] = (int)(8192u << 16);   // tensor_dim1 = 8192 (bits 111:80)
    g[3] = (int)(32u << 16);     // tile_dim0 = 32 (bits 127:112)
    g[4] = 128;                  // tile_dim1 = 128 (bits 143:128)
    g[5] = 4096;                 // tensor_dim0_stride = 4096 (bits 207:160)
    g[6] = 0;
    g[7] = 0;
    return g;
}

__device__ __forceinline__ void tdm_load(ull_t gaddr, uint_t lds_byte_off,
                                         tdm_v8i g1) {
    tdm_v4u g0;
    g0.x = 1u;                                       // count = 1 (valid)
    g0.y = lds_byte_off;                             // lds_addr
    g0.z = (uint_t)gaddr;                            // global_addr lo
    g0.w = (uint_t)(gaddr >> 32) | (2u << 30);       // global_addr hi | type=2
    tdm_v4i z4 = {0, 0, 0, 0};
#if TDM_6ARG
    tdm_v8i z8 = {0, 0, 0, 0, 0, 0, 0, 0};
    __builtin_amdgcn_tensor_load_to_lds(g0, g1, z4, z4, z8, 0);
#else
    __builtin_amdgcn_tensor_load_to_lds(g0, g1, z4, z4, 0);
#endif
}
#endif // HAVE_TDM

__global__ __launch_bounds__(256)
void k_gemm(const ushort_t* __restrict__ Xhi, const ushort_t* __restrict__ Xlo,
            const ushort_t* __restrict__ Whi, const ushort_t* __restrict__ Wlo,
            const float* __restrict__ bias, float* __restrict__ out) {
    __shared__ __attribute__((aligned(16))) ushort_t sXh[2][MT * LROW];
    __shared__ __attribute__((aligned(16))) ushort_t sXl[2][MT * LROW];
    __shared__ __attribute__((aligned(16))) ushort_t sWh[2][NT * LROW];
    __shared__ __attribute__((aligned(16))) ushort_t sWl[2][NT * LROW];

    const int m0 = blockIdx.y * MT;
    const int n0 = blockIdx.x * NT;
    const int t  = threadIdx.x;
    const int wave = __builtin_amdgcn_readfirstlane(t >> 5);
    const int lane = t & 31;
    const int wm = (wave >> 2) * 64;       // 2 wave-rows of 64
    const int wn = (wave & 3) * 32;        // 4 wave-cols of 32
    const int l16 = lane & 15, lh = lane >> 4;
    const int ac0 = lh * 8,  ac1 = lh * 8 + 16;   // A frag chunks (ISA 7.12.2)
    const int bc0 = lh * 8,  bc1 = lh * 8 + 16;   // B frag chunks (ISA 7.12.2)

    v8f acc[4][2];
    const v8f vzero = {0.f, 0.f, 0.f, 0.f, 0.f, 0.f, 0.f, 0.f};
    #pragma unroll
    for (int mi = 0; mi < 4; mi++)
        #pragma unroll
        for (int ni = 0; ni < 2; ni++) acc[mi][ni] = vzero;

#if HAVE_TDM
    // Wave 0 drives all four operand streams through the TDM.
    const bool issuer = (wave == 0);
    const ull_t gxh = (ull_t)(uintptr_t)Xhi + (ull_t)m0 * (DDIM * 2);
    const ull_t gxl = (ull_t)(uintptr_t)Xlo + (ull_t)m0 * (DDIM * 2);
    const ull_t gwh = (ull_t)(uintptr_t)Whi + (ull_t)n0 * (DDIM * 2);
    const ull_t gwl = (ull_t)(uintptr_t)Wlo + (ull_t)n0 * (DDIM * 2);
    const uint_t oxh[2] = {(uint_t)(uintptr_t)&sXh[0][0], (uint_t)(uintptr_t)&sXh[1][0]};
    const uint_t oxl[2] = {(uint_t)(uintptr_t)&sXl[0][0], (uint_t)(uintptr_t)&sXl[1][0]};
    const uint_t owh[2] = {(uint_t)(uintptr_t)&sWh[0][0], (uint_t)(uintptr_t)&sWh[1][0]};
    const uint_t owl[2] = {(uint_t)(uintptr_t)&sWl[0][0], (uint_t)(uintptr_t)&sWl[1][0]};
    const tdm_v8i g1 = tdm_make_g1();
    if (issuer) {                          // prologue: tiles for kt=0 -> buf0
        tdm_load(gxh, oxh[0], g1);
        tdm_load(gxl, oxl[0], g1);
        tdm_load(gwh, owh[0], g1);
        tdm_load(gwl, owl[0], g1);
    }
#else
    const int wseg = t & 3, wrow = t >> 2;        // 4 x uint4 chunks per row
#endif

    for (int kt = 0; kt < NK; ++kt) {
        const int buf = kt & 1;

#if HAVE_TDM
        if (kt + 1 < NK) {
            if (issuer) {
                const ull_t ko = (ull_t)(kt + 1) * (KT * 2);
                const int nb = (kt + 1) & 1;
                tdm_load(gxh + ko, oxh[nb], g1);
                tdm_load(gxl + ko, oxl[nb], g1);
                tdm_load(gwh + ko, owh[nb], g1);
                tdm_load(gwl + ko, owl[nb], g1);
                // in-order per wave: <=4 outstanding means kt's tiles landed
                __builtin_amdgcn_s_wait_tensorcnt((short)4);
            }
        } else if (issuer) {
            __builtin_amdgcn_s_wait_tensorcnt((short)0);
        }
        __syncthreads();
#else
        const int k0 = kt * KT;
        #pragma unroll
        for (int i = 0; i < 2; i++) {
            const int row = wrow + i * 64;
            const size_t xo = (size_t)(m0 + row) * DDIM + k0 + wseg * 8;
            const size_t wo = (size_t)(n0 + row) * DDIM + k0 + wseg * 8;
            *(uint4*)&sXh[buf][row * LROW + wseg * 8] = *(const uint4*)(Xhi + xo);
            *(uint4*)&sXl[buf][row * LROW + wseg * 8] = *(const uint4*)(Xlo + xo);
            *(uint4*)&sWh[buf][row * LROW + wseg * 8] = *(const uint4*)(Whi + wo);
            *(uint4*)&sWl[buf][row * LROW + wseg * 8] = *(const uint4*)(Wlo + wo);
        }
        __syncthreads();
#endif

        // ---- WMMA: 4x2 tiles, 3 products each (hi*hi + hi*lo + lo*hi)
        v16bf bh[2], bl[2];
        #pragma unroll
        for (int ni = 0; ni < 2; ni++) {
            const int r = wn + ni * 16 + l16;
            bh[ni] = ldfrag(&sWh[buf][0], r, bc0, bc1);
            bl[ni] = ldfrag(&sWl[buf][0], r, bc0, bc1);
        }
        #pragma unroll
        for (int mi = 0; mi < 4; mi++) {
            const int r = wm + mi * 16 + l16;
            const v16bf ah = ldfrag(&sXh[buf][0], r, ac0, ac1);
            const v16bf al = ldfrag(&sXl[buf][0], r, ac0, ac1);
            #pragma unroll
            for (int ni = 0; ni < 2; ni++) {
                acc[mi][ni] = __builtin_amdgcn_wmma_f32_16x16x32_bf16(
                    false, ah, false, bh[ni], (short)0, acc[mi][ni], false, false);
                acc[mi][ni] = __builtin_amdgcn_wmma_f32_16x16x32_bf16(
                    false, ah, false, bl[ni], (short)0, acc[mi][ni], false, false);
                acc[mi][ni] = __builtin_amdgcn_wmma_f32_16x16x32_bf16(
                    false, al, false, bh[ni], (short)0, acc[mi][ni], false, false);
            }
        }
        __syncthreads();   // tile fully consumed; next iter may overwrite
    }

    // ---- epilogue: C layout (ISA 7.12.2): lane%16 = n, m = vgpr + 8*(lane/16)
    #pragma unroll
    for (int ni = 0; ni < 2; ni++) {
        const int n = n0 + wn + ni * 16 + l16;
        const float bv = bias[n];
        #pragma unroll
        for (int mi = 0; mi < 4; mi++) {
            #pragma unroll
            for (int i = 0; i < 8; i++) {
                const int m = m0 + wm + mi * 16 + (i + 8 * lh);
                out[(size_t)m * DDIM + n] = acc[mi][ni][i] + bv;
            }
        }
    }
}

// ---------------------------------------------------------------- launcher
extern "C" void kernel_launch(void* const* d_in, const int* in_sizes, int n_in,
                              void* d_out, int out_size, void* d_ws, size_t ws_size,
                              hipStream_t stream) {
    const float* x     = (const float*)d_in[0];  // [4,2048,4096]
    const float* W     = (const float*)d_in[1];  // [4096,4096]
    const float* bias  = (const float*)d_in[2];  // [4096]
    const float* A     = (const float*)d_in[3];  // [8,8,4096]
    const float* Bm    = (const float*)d_in[4];  // [8,4096,8]
    const float* embed = (const float*)d_in[5];  // [1,4096]
    const float* rw    = (const float*)d_in[6];  // [8,4096]
    float* out = (float*)d_out;

    char* ws = (char*)d_ws;
    float*    gate = (float*)ws;
    ushort_t* whi  = (ushort_t*)(ws + 256);
    ushort_t* wlo  = (ushort_t*)(ws + 256 + (size_t)DDIM * DDIM * 2);
    ushort_t* xhi  = (ushort_t*)(ws + 256 + (size_t)DDIM * DDIM * 4);
    ushort_t* xlo  = (ushort_t*)(ws + 256 + (size_t)DDIM * DDIM * 4
                                        + (size_t)MDIM * DDIM * 2);

    k_router<<<1, 256, 0, stream>>>(embed, rw, gate);

    dim3 g2(DDIM / 64, DDIM / 64);
    k_weff<<<g2, 256, 0, stream>>>(W, Bm, A, gate, whi, wlo);

    const int nxs = (int)(((size_t)MDIM * DDIM / 8) / 256);
    k_xsplit<<<nxs, 256, 0, stream>>>(x, xhi, xlo);

    dim3 g3(DDIM / NT, MDIM / MT);
    k_gemm<<<g3, 256, 0, stream>>>(xhi, xlo, whi, wlo, bias, out);
}
